// DEC_59880434041064
// MI455X (gfx1250) — compile-verified, matrix operand order checked by reading, not run on hardware
//
#include <hip/hip_runtime.h>

// ---------------------------------------------------------------------------
// Problem constants (from the reference file)
// ---------------------------------------------------------------------------
#define B_ 1024
#define L_ 512
#define H_ 128
#define G3_ 384          // 3*H
#define ATTN_ 5
#define DLY_ 10
#define LM5_ (L_ - ATTN_)   // 507
#define NBLK_ (B_ / 16)     // 64 batch tiles per GRU

typedef __attribute__((ext_vector_type(16))) __bf16 v16bf;
typedef __attribute__((ext_vector_type(8)))  float  v8f;

union BfFrag { v16bf v; unsigned short u[16]; };

__device__ __forceinline__ unsigned short f32_to_bf16(float f) {
    unsigned int x = __builtin_bit_cast(unsigned int, f);
    unsigned int r = x + 0x7FFFu + ((x >> 16) & 1u);
    return (unsigned short)(r >> 16);
}
__device__ __forceinline__ float bf16lo_to_f32(unsigned int u) {
    return __builtin_bit_cast(float, u << 16);
}
__device__ __forceinline__ float bf16hi_to_f32(unsigned int u) {
    return __builtin_bit_cast(float, u & 0xFFFF0000u);
}

// ---------------------------------------------------------------------------
// K0: f32 -> bf16 weight conversion
// ---------------------------------------------------------------------------
__global__ void cvt_bf16_kernel(const float* __restrict__ src,
                                unsigned short* __restrict__ dst, int n) {
    int i = blockIdx.x * blockDim.x + threadIdx.x;
    if (i < n) dst[i] = f32_to_bf16(src[i]);
}

// ---------------------------------------------------------------------------
// K1: layer-0 input gates  gi0[t*B+b, :] = x[b,t,:] @ w_ih.T + b_ih   (K=2)
// ---------------------------------------------------------------------------
__global__ void gi_l0_kernel(const float* __restrict__ x,
                             const float* __restrict__ w_ih,   // [384,2]
                             const float* __restrict__ b_ih,   // [384]
                             unsigned short* __restrict__ gi) {
    long idx = (long)blockIdx.x * blockDim.x + threadIdx.x;
    const long total = (long)L_ * B_ * G3_;
    if (idx >= total) return;
    int  c = (int)(idx % G3_);
    long m = idx / G3_;
    int  b = (int)(m % B_);
    int  t = (int)(m / B_);
    float x0 = x[((long)b * L_ + t) * 2 + 0];
    float x1 = x[((long)b * L_ + t) * 2 + 1];
    float v  = x0 * w_ih[c * 2 + 0] + x1 * w_ih[c * 2 + 1] + b_ih[c];
    gi[idx] = f32_to_bf16(v);
}

// ---------------------------------------------------------------------------
// GRU recurrent body.  One block = 16 batch rows, loops over all timesteps.
// 24 waves; wave w owns output columns [16w,16w+16) of the 384-wide gh.
// w_hh B-fragments persistent in VGPRs; h in LDS as bf16; gi(t) prefetched
// into registers before the step's first barrier.
// MODE 0: h -> bf16 out_bf [t*B+b,128];  MODE 1: h -> f32 out_f32 [b*L+t,128]
// ---------------------------------------------------------------------------
template <int MODE>
__device__ __forceinline__ void gru_body(
        const unsigned short* __restrict__ gi,      // [L*B, 384] bf16
        const unsigned short* __restrict__ whh_bf,  // [384,128] bf16
        const float* __restrict__ b_hh,             // [384]
        unsigned short* __restrict__ out_bf,
        float* __restrict__ out_f32,
        int b0) {
    __shared__ unsigned short h_bf[16][136];   // padded rows -> distinct banks
    __shared__ float gh[16][385];
    __shared__ float bhh_s[G3_];

    const int tid  = threadIdx.x;
    const int wv   = tid >> 5;
    const int lane = tid & 31;

    for (int i = tid; i < G3_; i += 768) bhh_s[i] = b_hh[i];
    for (int i = tid; i < 16 * 136; i += 768) ((unsigned short*)h_bf)[i] = 0;

    const int n0    = wv * 16;
    const int col   = n0 + (lane & 15);
    const int kbase = (lane < 16) ? 0 : 8;
    const int arow  = lane & 15;

    // Persistent B fragments: B[k][n] = w_hh[n][k]
    BfFrag bfrag[4];
#pragma unroll
    for (int kc = 0; kc < 4; ++kc) {
#pragma unroll
        for (int j = 0; j < 16; ++j) {
            int k = kc * 32 + kbase + (j < 8 ? j : j + 8);
            bfrag[kc].u[j] = whh_bf[(long)col * 128 + k];
        }
    }

    // Gate-update decomposition: 16 rows x 64 bf16-pairs = 1024 pairs.
    const int p0_row = tid >> 6,           p0_c = (tid & 63) * 2;
    const int p1     = tid + 768;
    const int p1_row = p1 >> 6,            p1_c = (p1 & 63) * 2;
    const bool has2  = (tid < 256);

    for (int t = 0; t < L_; ++t) {
        // ---- prefetch gi(t) into registers (independent of h) ----
        unsigned int g0r, g0z, g0n, g1r = 0, g1z = 0, g1n = 0;
        {
            const unsigned short* g = gi + ((long)t * B_ + (b0 + p0_row)) * G3_;
            g0r = *(const unsigned int*)(g + p0_c);
            g0z = *(const unsigned int*)(g + 128 + p0_c);
            g0n = *(const unsigned int*)(g + 256 + p0_c);
        }
        if (has2) {
            const unsigned short* g = gi + ((long)t * B_ + (b0 + p1_row)) * G3_;
            g1r = *(const unsigned int*)(g + p1_c);
            g1z = *(const unsigned int*)(g + 128 + p1_c);
            g1n = *(const unsigned int*)(g + 256 + p1_c);
        }

        __syncthreads();             // h_bf ready (init or previous step)

        // ---- WMMA phase: load all A chunks first, then 4 chained WMMAs ----
        BfFrag a[4];
#pragma unroll
        for (int kc = 0; kc < 4; ++kc) {
#pragma unroll
            for (int j = 0; j < 16; ++j) {
                int k = kc * 32 + kbase + (j < 8 ? j : j + 8);
                a[kc].u[j] = h_bf[arow][k];
            }
        }
        v8f acc = {};
#pragma unroll
        for (int kc = 0; kc < 4; ++kc)
            acc = __builtin_amdgcn_wmma_f32_16x16x32_bf16(
                false, a[kc].v, false, bfrag[kc].v, (short)0, acc, false, false);

#pragma unroll
        for (int v = 0; v < 8; ++v) {
            int m = v + ((lane < 16) ? 0 : 8);
            gh[m][col] = acc[v];
        }
        __syncthreads();             // gh tile complete

        // ---- fused GRU gate update, two bf16 lanes (c, c+1) per pair ----
#pragma unroll
        for (int pp = 0; pp < 2; ++pp) {
            if (pp == 1 && !has2) break;
            const int row = pp ? p1_row : p0_row;
            const int c   = pp ? p1_c   : p0_c;
            const unsigned int gr = pp ? g1r : g0r;
            const unsigned int gz = pp ? g1z : g0z;
            const unsigned int gn = pp ? g1n : g0n;
            const int bg = b0 + row;

            unsigned int hpair = *(const unsigned int*)&h_bf[row][c];
            float h0 = bf16lo_to_f32(hpair), h1 = bf16hi_to_f32(hpair);

            float hr0 = gh[row][c]       + bhh_s[c];
            float hr1 = gh[row][c + 1]   + bhh_s[c + 1];
            float hz0 = gh[row][128 + c] + bhh_s[128 + c];
            float hz1 = gh[row][129 + c] + bhh_s[129 + c];
            float hn0 = gh[row][256 + c] + bhh_s[256 + c];
            float hn1 = gh[row][257 + c] + bhh_s[257 + c];

            float r0 = 1.f / (1.f + __expf(-(bf16lo_to_f32(gr) + hr0)));
            float r1 = 1.f / (1.f + __expf(-(bf16hi_to_f32(gr) + hr1)));
            float z0 = 1.f / (1.f + __expf(-(bf16lo_to_f32(gz) + hz0)));
            float z1 = 1.f / (1.f + __expf(-(bf16hi_to_f32(gz) + hz1)));
            float n0v = tanhf(bf16lo_to_f32(gn) + r0 * hn0);
            float n1v = tanhf(bf16hi_to_f32(gn) + r1 * hn1);
            float hnew0 = (1.f - z0) * n0v + z0 * h0;
            float hnew1 = (1.f - z1) * n1v + z1 * h1;

            unsigned int packed =
                (unsigned int)f32_to_bf16(hnew0) |
                ((unsigned int)f32_to_bf16(hnew1) << 16);
            *(unsigned int*)&h_bf[row][c] = packed;
            if (MODE == 0) {
                *(unsigned int*)&out_bf[((long)t * B_ + bg) * H_ + c] = packed;
            } else {
                *(float2*)&out_f32[((long)bg * L_ + t) * H_ + c] =
                    make_float2(hnew0, hnew1);
            }
        }
    }
}

template <int MODE>
__global__ __launch_bounds__(768) void gru_recurrent_kernel(
        const unsigned short* __restrict__ gi,
        const unsigned short* __restrict__ whh_bf,
        const float* __restrict__ b_hh,
        unsigned short* __restrict__ out_bf,
        float* __restrict__ out_f32) {
    gru_body<MODE>(gi, whh_bf, b_hh, out_bf, out_f32, blockIdx.x * 16);
}

// Dual-GRU launch: blocks [0,64) run GRU-A, [64,128) run GRU-B concurrently.
template <int MODE>
__global__ __launch_bounds__(768) void gru_recurrent_dual_kernel(
        const unsigned short* __restrict__ giA,
        const unsigned short* __restrict__ whhA,
        const float* __restrict__ bhhA,
        unsigned short* __restrict__ outbfA, float* __restrict__ outf32A,
        const unsigned short* __restrict__ giB,
        const unsigned short* __restrict__ whhB,
        const float* __restrict__ bhhB,
        unsigned short* __restrict__ outbfB, float* __restrict__ outf32B) {
    if (blockIdx.x < NBLK_)
        gru_body<MODE>(giA, whhA, bhhA, outbfA, outf32A, blockIdx.x * 16);
    else
        gru_body<MODE>(giB, whhB, bhhB, outbfB, outf32B,
                       (blockIdx.x - NBLK_) * 16);
}

// ---------------------------------------------------------------------------
// K3: WMMA GEMM, K=128:  Out[m,n] = A[m,:] . W[n,:] + bias[n]   (bf16 out)
//     Each wave: one 16-row M tile x 4 N tiles, A fragments held in registers.
// ---------------------------------------------------------------------------
__global__ __launch_bounds__(256) void gemm_k128_kernel(
        const unsigned short* __restrict__ A,
        const unsigned short* __restrict__ W,
        const float* __restrict__ bias,
        unsigned short* __restrict__ Out,
        int Mtiles) {
    const int WPM = (G3_ / 16) / 4;               // 6 waves per M tile
    const int wv = threadIdx.x >> 5, lane = threadIdx.x & 31;
    long gw = (long)blockIdx.x * 8 + wv;
    long tm = gw / WPM;
    int tn0 = (int)(gw % WPM) * 4;
    if (tm >= Mtiles) return;
    const int m0    = (int)tm * 16;
    const int kbase = (lane < 16) ? 0 : 8;
    const int arow  = m0 + (lane & 15);

    BfFrag a[4];
#pragma unroll
    for (int kc = 0; kc < 4; ++kc)
#pragma unroll
        for (int j = 0; j < 16; ++j) {
            int k = kc * 32 + kbase + (j < 8 ? j : j + 8);
            a[kc].u[j] = A[(long)arow * 128 + k];
        }

#pragma unroll
    for (int i = 0; i < 4; ++i) {
        const int col = (tn0 + i) * 16 + (lane & 15);
        v8f acc = {};
#pragma unroll
        for (int kc = 0; kc < 4; ++kc) {
            BfFrag bm;
#pragma unroll
            for (int j = 0; j < 16; ++j) {
                int k = kc * 32 + kbase + (j < 8 ? j : j + 8);
                bm.u[j] = W[(long)col * 128 + k];
            }
            acc = __builtin_amdgcn_wmma_f32_16x16x32_bf16(
                false, a[kc].v, false, bm.v, (short)0, acc, false, false);
        }
        float bv = bias[col];
#pragma unroll
        for (int v = 0; v < 8; ++v) {
            long m = m0 + v + ((lane < 16) ? 0 : 8);
            Out[m * G3_ + col] = f32_to_bf16(acc[v] + bv);
        }
    }
}

// ---------------------------------------------------------------------------
// K4: attention scores  sh[b,l] = r[b,l,:].w_a[:H]; sp = r.w_a[H:]
// ---------------------------------------------------------------------------
__global__ void attn_scores_kernel(const float* __restrict__ r,
                                   const float* __restrict__ w_a,
                                   float* __restrict__ sh,
                                   float* __restrict__ sp) {
    long idx = (long)blockIdx.x * blockDim.x + threadIdx.x;
    if (idx >= (long)B_ * L_) return;
    const float4* row = (const float4*)(r + idx * H_);
    const float4* wa  = (const float4*)w_a;
    float s0 = 0.f, s1 = 0.f;
#pragma unroll 8
    for (int k = 0; k < H_ / 4; ++k) {
        float4 v = row[k], a = wa[k], b = wa[H_ / 4 + k];
        s0 += v.x * a.x + v.y * a.y + v.z * a.z + v.w * a.w;
        s1 += v.x * b.x + v.y * b.y + v.z * b.z + v.w * b.w;
    }
    sh[idx] = s0;
    sp[idx] = s1;
}

// ---------------------------------------------------------------------------
// K5: softmax over the 5-window + build ccat[m,0:128]=context, [128:256]=r_t
// ---------------------------------------------------------------------------
__global__ void attn_ctx_kernel(const float* __restrict__ r,
                                const float* __restrict__ sh,
                                const float* __restrict__ sp,
                                const float* __restrict__ b_a,
                                unsigned short* __restrict__ ccat) {
    long gid = (long)blockIdx.x * blockDim.x + threadIdx.x;
    const long total = (long)B_ * LM5_ * H_;
    if (gid >= total) return;
    int  c = (int)(gid & 127);
    long m = gid >> 7;
    int  i = (int)(m % LM5_);
    int  b = (int)(m / LM5_);
    float ba  = b_a[0];
    float shv = sh[(long)b * L_ + i + ATTN_];
    float e[ATTN_], emax = -1e30f;
#pragma unroll
    for (int j = 0; j < ATTN_; ++j) {
        e[j] = sp[(long)b * L_ + i + j] + shv + ba;
        emax = fmaxf(emax, e[j]);
    }
    float den = 0.f;
#pragma unroll
    for (int j = 0; j < ATTN_; ++j) { e[j] = __expf(e[j] - emax); den += e[j]; }
    float inv = 1.f / den;
    float cv = 0.f;
#pragma unroll
    for (int j = 0; j < ATTN_; ++j)
        cv += e[j] * inv * r[((long)b * L_ + i + j) * H_ + c];
    ccat[m * 256 + c]       = f32_to_bf16(cv);
    ccat[m * 256 + 128 + c] = f32_to_bf16(r[((long)b * L_ + i + ATTN_) * H_ + c]);
}

// ---------------------------------------------------------------------------
// K6: WMMA GEMM, K=256:  new = ccat @ w_c.T + b_c, scattered IN PLACE into
//     r[b, ATTN+i, :].  Each wave: one M tile x 4 N tiles, A in registers.
// ---------------------------------------------------------------------------
__global__ __launch_bounds__(256) void gemm_k256_kernel(
        const unsigned short* __restrict__ A,   // [M,256] ccat
        const unsigned short* __restrict__ W,   // [128,256] w_c bf16
        const float* __restrict__ bias,         // [128]
        float* __restrict__ Out,                // r buffer [B,L,128]
        int Mtiles) {
    const int WPM = (H_ / 16) / 4;              // 2 waves per M tile
    const int wv = threadIdx.x >> 5, lane = threadIdx.x & 31;
    long gw = (long)blockIdx.x * 8 + wv;
    long tm = gw / WPM;
    int tn0 = (int)(gw % WPM) * 4;
    if (tm >= Mtiles) return;
    const int m0    = (int)tm * 16;
    const int kbase = (lane < 16) ? 0 : 8;
    const int arow  = m0 + (lane & 15);

    BfFrag a[8];
#pragma unroll
    for (int kc = 0; kc < 8; ++kc)
#pragma unroll
        for (int j = 0; j < 16; ++j) {
            int k = kc * 32 + kbase + (j < 8 ? j : j + 8);
            a[kc].u[j] = A[(long)arow * 256 + k];
        }

#pragma unroll
    for (int i = 0; i < 4; ++i) {
        const int col = (tn0 + i) * 16 + (lane & 15);
        v8f acc = {};
#pragma unroll
        for (int kc = 0; kc < 8; ++kc) {
            BfFrag bm;
#pragma unroll
            for (int j = 0; j < 16; ++j) {
                int k = kc * 32 + kbase + (j < 8 ? j : j + 8);
                bm.u[j] = W[(long)col * 256 + k];
            }
            acc = __builtin_amdgcn_wmma_f32_16x16x32_bf16(
                false, a[kc].v, false, bm.v, (short)0, acc, false, false);
        }
        float bv = bias[col];
#pragma unroll
        for (int v = 0; v < 8; ++v) {
            long m = m0 + v + ((lane < 16) ? 0 : 8);
            int  b  = (int)(m / LM5_);
            int  ii = (int)(m % LM5_);
            Out[((long)b * L_ + ATTN_ + ii) * H_ + col] = acc[v] + bv;
        }
    }
}

// ---------------------------------------------------------------------------
// K7: final   out[b,t] = sigmoid(c1[b,t].w_o[:H] + c2[b,min(t+DLY,L-1)].w_o[H:] + b_o)
// ---------------------------------------------------------------------------
__global__ void final_kernel(const float* __restrict__ c1,
                             const float* __restrict__ c2,
                             const float* __restrict__ w_o,
                             const float* __restrict__ b_o,
                             float* __restrict__ out) {
    long idx = (long)blockIdx.x * blockDim.x + threadIdx.x;
    if (idx >= (long)B_ * L_) return;
    int  t = (int)(idx % L_);
    long b = idx / L_;
    int  td = t + DLY_; if (td > L_ - 1) td = L_ - 1;
    const float4* p1 = (const float4*)(c1 + idx * H_);
    const float4* p2 = (const float4*)(c2 + ((long)b * L_ + td) * H_);
    const float4* wo = (const float4*)w_o;
    float s = b_o[0];
#pragma unroll 8
    for (int k = 0; k < H_ / 4; ++k) {
        float4 u = p1[k], v = p2[k], a = wo[k], bq = wo[H_ / 4 + k];
        s += u.x * a.x + u.y * a.y + u.z * a.z + u.w * a.w;
        s += v.x * bq.x + v.y * bq.y + v.z * bq.z + v.w * bq.w;
    }
    out[idx] = 1.f / (1.f + __expf(-s));
}

// ---------------------------------------------------------------------------
// Host launcher
// ---------------------------------------------------------------------------
extern "C" void kernel_launch(void* const* d_in, const int* in_sizes, int n_in,
                              void* d_out, int out_size, void* d_ws, size_t ws_size,
                              hipStream_t stream) {
    (void)in_sizes; (void)n_in; (void)out_size;

    const float* received = (const float*)d_in[0];
    const float* w_ih1_l0 = (const float*)d_in[1];
    const float* w_hh1_l0 = (const float*)d_in[2];
    const float* b_ih1_l0 = (const float*)d_in[3];
    const float* b_hh1_l0 = (const float*)d_in[4];
    const float* w_ih1_l1 = (const float*)d_in[5];
    const float* w_hh1_l1 = (const float*)d_in[6];
    const float* b_ih1_l1 = (const float*)d_in[7];
    const float* b_hh1_l1 = (const float*)d_in[8];
    const float* w_ih2_l0 = (const float*)d_in[9];
    const float* w_hh2_l0 = (const float*)d_in[10];
    const float* b_ih2_l0 = (const float*)d_in[11];
    const float* b_hh2_l0 = (const float*)d_in[12];
    const float* w_ih2_l1 = (const float*)d_in[13];
    const float* w_hh2_l1 = (const float*)d_in[14];
    const float* b_ih2_l1 = (const float*)d_in[15];
    const float* b_hh2_l1 = (const float*)d_in[16];
    const float* w_a      = (const float*)d_in[17];
    const float* b_a      = (const float*)d_in[18];
    const float* w_c      = (const float*)d_in[19];
    const float* b_c      = (const float*)d_in[20];
    const float* w_o      = (const float*)d_in[21];
    const float* b_o      = (const float*)d_in[22];
    float* out = (float*)d_out;

    const size_t szGi  = (size_t)L_ * B_ * G3_ * 2;   // 402,653,184
    const size_t szRl0 = (size_t)L_ * B_ * H_ * 2;    // 134,217,728
    const size_t szR   = (size_t)B_ * L_ * H_ * 4;    // 268,435,456
    const size_t szW   = (size_t)G3_ * H_ * 2;        //      98,304
    const size_t szWc  = (size_t)H_ * 2 * H_ * 2;     //      65,536
    const size_t needDual   = 2 * szGi + 2 * szRl0 + 2 * szR + 6 * szW + szWc;
    const size_t needSerial =     szGi +     szRl0 + 2 * szR + 6 * szW + szWc;
    const bool dual = (ws_size >= needDual);

    char* ws = (char*)d_ws;
    size_t off = 0;
    auto take = [&](size_t bytes) -> char* {
        char* p = ws + off;
        off += (bytes + 255) & ~(size_t)255;
        return p;
    };
    (void)needSerial;
    unsigned short* giA  = (unsigned short*)take(szGi);           // alias: ccat
    unsigned short* giB  = dual ? (unsigned short*)take(szGi) : giA;
    unsigned short* rl0A = (unsigned short*)take(szRl0);          // alias: sh,sp
    unsigned short* rl0B = dual ? (unsigned short*)take(szRl0) : rl0A;
    float* r1 = (float*)take(szR);
    float* r2 = (float*)take(szR);
    unsigned short* whh1_0 = (unsigned short*)take(szW);
    unsigned short* whh1_1 = (unsigned short*)take(szW);
    unsigned short* whh2_0 = (unsigned short*)take(szW);
    unsigned short* whh2_1 = (unsigned short*)take(szW);
    unsigned short* wih1_1 = (unsigned short*)take(szW);
    unsigned short* wih2_1 = (unsigned short*)take(szW);
    unsigned short* wc_bf  = (unsigned short*)take(szWc);
    unsigned short* ccat = giA;
    float* sh = (float*)rl0A;
    float* sp = (float*)((char*)rl0A + (size_t)B_ * L_ * 4);

    // --- K0: weight conversion ---
    const int wn = G3_ * H_;
    cvt_bf16_kernel<<<(wn + 255) / 256, 256, 0, stream>>>(w_hh1_l0, whh1_0, wn);
    cvt_bf16_kernel<<<(wn + 255) / 256, 256, 0, stream>>>(w_hh1_l1, whh1_1, wn);
    cvt_bf16_kernel<<<(wn + 255) / 256, 256, 0, stream>>>(w_hh2_l0, whh2_0, wn);
    cvt_bf16_kernel<<<(wn + 255) / 256, 256, 0, stream>>>(w_hh2_l1, whh2_1, wn);
    cvt_bf16_kernel<<<(wn + 255) / 256, 256, 0, stream>>>(w_ih1_l1, wih1_1, wn);
    cvt_bf16_kernel<<<(wn + 255) / 256, 256, 0, stream>>>(w_ih2_l1, wih2_1, wn);
    const int wcn = H_ * 2 * H_;
    cvt_bf16_kernel<<<(wcn + 255) / 256, 256, 0, stream>>>(w_c, wc_bf, wcn);

    const long giTotal = (long)L_ * B_ * G3_;
    const int giBlocks = (int)((giTotal + 255) / 256);
    const int Mtiles_gi = (L_ * B_) / 16;                 // 32768
    const int giGemmBlocks = Mtiles_gi * 6 / 8;           // 6 waves per M tile

    if (dual) {
        // Both GRUs' independent scans run concurrently (grid = 128).
        gi_l0_kernel<<<giBlocks, 256, 0, stream>>>(received, w_ih1_l0, b_ih1_l0, giA);
        gi_l0_kernel<<<giBlocks, 256, 0, stream>>>(received, w_ih2_l0, b_ih2_l0, giB);
        gru_recurrent_dual_kernel<0><<<2 * NBLK_, 768, 0, stream>>>(
            giA, whh1_0, b_hh1_l0, rl0A, nullptr,
            giB, whh2_0, b_hh2_l0, rl0B, nullptr);
        gemm_k128_kernel<<<giGemmBlocks, 256, 0, stream>>>(rl0A, wih1_1, b_ih1_l1, giA, Mtiles_gi);
        gemm_k128_kernel<<<giGemmBlocks, 256, 0, stream>>>(rl0B, wih2_1, b_ih2_l1, giB, Mtiles_gi);
        gru_recurrent_dual_kernel<1><<<2 * NBLK_, 768, 0, stream>>>(
            giA, whh1_1, b_hh1_l1, nullptr, r1,
            giB, whh2_1, b_hh2_l1, nullptr, r2);
    } else {
        // Serialized fallback with aliased single gi / rl0 buffers.
        gi_l0_kernel<<<giBlocks, 256, 0, stream>>>(received, w_ih1_l0, b_ih1_l0, giA);
        gru_recurrent_kernel<0><<<NBLK_, 768, 0, stream>>>(giA, whh1_0, b_hh1_l0, rl0A, nullptr);
        gemm_k128_kernel<<<giGemmBlocks, 256, 0, stream>>>(rl0A, wih1_1, b_ih1_l1, giA, Mtiles_gi);
        gru_recurrent_kernel<1><<<NBLK_, 768, 0, stream>>>(giA, whh1_1, b_hh1_l1, nullptr, r1);

        gi_l0_kernel<<<giBlocks, 256, 0, stream>>>(received, w_ih2_l0, b_ih2_l0, giA);
        gru_recurrent_kernel<0><<<NBLK_, 768, 0, stream>>>(giA, whh2_0, b_hh2_l0, rl0A, nullptr);
        gemm_k128_kernel<<<giGemmBlocks, 256, 0, stream>>>(rl0A, wih2_1, b_ih2_l1, giA, Mtiles_gi);
        gru_recurrent_kernel<1><<<NBLK_, 768, 0, stream>>>(giA, whh2_1, b_hh2_l1, nullptr, r2);
    }

    // --- Attention (in-place: r becomes concat([r[:, :a], new]) ) ---
    const int scBlocks  = (int)(((long)B_ * L_ + 255) / 256);
    const int ctxBlocks = (int)(((long)B_ * LM5_ * H_ + 255) / 256);
    const int Mtiles_c  = (B_ * LM5_) / 16;               // 32448
    const int cGemmBlocks = Mtiles_c * 2 / 8;             // 2 waves per M tile

    attn_scores_kernel<<<scBlocks, 256, 0, stream>>>(r1, w_a, sh, sp);
    attn_ctx_kernel<<<ctxBlocks, 256, 0, stream>>>(r1, sh, sp, b_a, ccat);
    gemm_k256_kernel<<<cGemmBlocks, 256, 0, stream>>>(ccat, wc_bf, b_c, r1, Mtiles_c);

    attn_scores_kernel<<<scBlocks, 256, 0, stream>>>(r2, w_a, sh, sp);
    attn_ctx_kernel<<<ctxBlocks, 256, 0, stream>>>(r2, sh, sp, b_a, ccat);
    gemm_k256_kernel<<<cGemmBlocks, 256, 0, stream>>>(ccat, wc_bf, b_c, r2, Mtiles_c);

    // --- Final output ---
    final_kernel<<<scBlocks, 256, 0, stream>>>(r1, r2, w_o, b_o, out);
}
